// ConvLayer_20839181320724
// MI455X (gfx1250) — matrix operand order repaired and checked by reading, not compiled, not run on hardware
//
#include <hip/hip_runtime.h>
#include <hip/hip_bf16.h>
#include <math.h>

typedef __bf16 bf16;
typedef __attribute__((ext_vector_type(16))) __bf16 v16bf;
typedef __attribute__((ext_vector_type(8)))  __bf16 v8bf;
typedef __attribute__((ext_vector_type(8)))  float  v8f;
typedef __attribute__((ext_vector_type(4)))  float  v4f;

#define N_NODES 20000
#define N_EDGES 320000

// ---------- CDNA5 helpers ----------

__device__ __forceinline__ void lds_fence() {
  // intra-wave LDS RAW fence (wave32 lockstep; DS ops in-order per wave)
  asm volatile("s_wait_dscnt 0" ::: "memory");
}

__device__ __forceinline__ v8f wmma_bf16(v16bf a, v16bf b, v8f c) {
  return __builtin_amdgcn_wmma_f32_16x16x32_bf16(false, a, false, b, (short)0, c,
                                                 false, false);
}

union V16U {
  v16bf v;
  v8bf h8[2];
};

// A fragment (16x32 bf16) from a contiguous bf16 row (LDS or global):
// lane L holds row r=L&15 (caller passes that row's pointer); half h=L>>4;
// elements 0..7 -> K = kc+h*8.. (contiguous), 8..15 -> K = kc+16+h*8.. (contiguous)
__device__ __forceinline__ v16bf load_a_bf16row(const bf16* row, int kc) {
  const int h = (threadIdx.x >> 4) & 1;
  V16U u;
  u.h8[0] = *(const v8bf*)(row + kc + h * 8);
  u.h8[1] = *(const v8bf*)(row + kc + 16 + h * 8);
  return u.v;
}

// A fragment from a contiguous f32 row (vector loads + convert)
__device__ __forceinline__ v16bf load_a_f32row(const float* row, int kc) {
  const int h = (threadIdx.x >> 4) & 1;
  v4f a0 = *(const v4f*)(row + kc + h * 8);
  v4f a1 = *(const v4f*)(row + kc + h * 8 + 4);
  v4f b0 = *(const v4f*)(row + kc + 16 + h * 8);
  v4f b1 = *(const v4f*)(row + kc + 16 + h * 8 + 4);
  v16bf a;
#pragma unroll
  for (int e = 0; e < 4; ++e) {
    a[e] = (bf16)a0[e];
    a[4 + e] = (bf16)a1[e];
    a[8 + e] = (bf16)b0[e];
    a[12 + e] = (bf16)b1[e];
  }
  return a;
}

// A fragment = (rowA + rowB) elementwise, both contiguous f32 rows
__device__ __forceinline__ v16bf load_a_f32sum(const float* rowA, const float* rowB,
                                               int kc) {
  const int h = (threadIdx.x >> 4) & 1;
  v4f a0 = *(const v4f*)(rowA + kc + h * 8);
  v4f a1 = *(const v4f*)(rowA + kc + h * 8 + 4);
  v4f b0 = *(const v4f*)(rowA + kc + 16 + h * 8);
  v4f b1 = *(const v4f*)(rowA + kc + 16 + h * 8 + 4);
  v4f c0 = *(const v4f*)(rowB + kc + h * 8);
  v4f c1 = *(const v4f*)(rowB + kc + h * 8 + 4);
  v4f d0 = *(const v4f*)(rowB + kc + 16 + h * 8);
  v4f d1 = *(const v4f*)(rowB + kc + 16 + h * 8 + 4);
  v16bf a;
#pragma unroll
  for (int e = 0; e < 4; ++e) {
    a[e] = (bf16)(a0[e] + c0[e]);
    a[4 + e] = (bf16)(a1[e] + c1[e]);
    a[8 + e] = (bf16)(b0[e] + d0[e]);
    a[12 + e] = (bf16)(b1[e] + d1[e]);
  }
  return a;
}

// 24 contiguous floats as 6x b128 loads (serves the stride-3 vector features:
// elements k..k+7 of all three components i=0..2 live in one 24-float span)
union Span24 {
  v4f v[6];
  float f[24];
};
__device__ __forceinline__ void load_span24(const float* p, Span24& s) {
#pragma unroll
  for (int q = 0; q < 6; ++q) s.v[q] = *(const v4f*)(p + 4 * q);
}

// Packed B fragment: weights pre-permuted to [tile][kchunk][lane][16] order,
// so each lane loads 32 contiguous bytes (2x global_load_b128).
__device__ __forceinline__ v16bf load_bp(const bf16* __restrict__ Wp, int chunks,
                                         int c, int t) {
  const v16bf* p = (const v16bf*)Wp;
  return p[(size_t)(t * chunks + c) * 32 + (threadIdx.x & 31)];
}

__device__ __forceinline__ float sspf(float x) {
  return fmaxf(x, 0.f) + log1pf(expf(-fabsf(x))) - 0.69314718055994531f;
}
__device__ __forceinline__ float siluf(float x) { return x / (1.f + expf(-x)); }

// ---------- weight prep: fold norm consts, convert to bf16, pack B-fragments ----------

__global__ void pack_b(bf16* __restrict__ dst, const float* __restrict__ src, int K,
                       int N, float s) {
  int idx = blockIdx.x * 256 + threadIdx.x;
  if (idx >= K * N) return;
  int e = idx & 15;
  int lane = (idx >> 4) & 31;
  int fi = idx >> 9;
  int chunks = K >> 5;
  int c = fi % chunks;
  int t = fi / chunks;
  int n = t * 16 + (lane & 15);
  int k = c * 32 + (lane >> 4) * 8 + (e & 7) + ((e >> 3) << 4);
  dst[idx] = (bf16)(src[k * N + n] * s);
}

// ---------- node kernel: pre_x, gate, xg(self_x) ----------

__global__ void __launch_bounds__(64) node_kernel(
    const float* __restrict__ x,
    const bf16* __restrict__ Wpre0, const float* __restrict__ bpre0,
    const bf16* __restrict__ Wpre1,
    const bf16* __restrict__ Wg1, const float* __restrict__ bg1,
    const bf16* __restrict__ Wg2, const float* __restrict__ bg2,
    const bf16* __restrict__ Wn0, const float* __restrict__ bn0,
    const bf16* __restrict__ Wn1,
    bf16* __restrict__ ps, bf16* __restrict__ pv,
    float* __restrict__ gs, float* __restrict__ gv) {
  __shared__ alignas(16) bf16  sb[2][16][128];  // f0 staging, then g1 staging
  __shared__ alignas(16) float sg[2][16][128];  // g (f32)
  const int wv = threadIdx.x >> 5;
  const int lane = threadIdx.x & 31;
  const int tile0 = (blockIdx.x * 2 + wv) * 16;
  const int nlo = lane & 15, h = lane >> 4;
  const float* xrow = x + (size_t)(tile0 + nlo) * 256;

  // ---- pre_x scalar: ps = xs @ (Wpre0/8) + b0 ----
  v16bf xs0 = load_a_f32row(xrow, 0);
  v16bf xs1 = load_a_f32row(xrow, 32);
#pragma unroll
  for (int t = 0; t < 4; ++t) {
    v8f acc = {};
    acc = wmma_bf16(xs0, load_bp(Wpre0, 2, 0, t), acc);
    acc = wmma_bf16(xs1, load_bp(Wpre0, 2, 1, t), acc);
    float bv = bpre0[t * 16 + nlo];
#pragma unroll
    for (int rr = 0; rr < 8; ++rr)
      ps[(size_t)(tile0 + h * 8 + rr) * 64 + t * 16 + nlo] = (bf16)(acc[rr] + bv);
  }

  // ---- pre_x vector: pv[:,:,i] = xv[:,:,i] @ (Wpre1/8) ----
  {
    v16bf af[3][2];
    const float* xv = xrow + 64;
#pragma unroll
    for (int kc2 = 0; kc2 < 2; ++kc2) {
      const int kc = kc2 * 32;
      Span24 lo, hi;
      load_span24(xv + 3 * (kc + h * 8), lo);
      load_span24(xv + 3 * (kc + 16 + h * 8), hi);
#pragma unroll
      for (int i = 0; i < 3; ++i)
#pragma unroll
        for (int e = 0; e < 8; ++e) {
          af[i][kc2][e] = (bf16)lo.f[3 * e + i];
          af[i][kc2][8 + e] = (bf16)hi.f[3 * e + i];
        }
    }
#pragma unroll 1
    for (int i = 0; i < 3; ++i) {
#pragma unroll
      for (int t = 0; t < 4; ++t) {
        v8f acc = {};
        acc = wmma_bf16(af[i][0], load_bp(Wpre1, 2, 0, t), acc);
        acc = wmma_bf16(af[i][1], load_bp(Wpre1, 2, 1, t), acc);
#pragma unroll
        for (int rr = 0; rr < 8; ++rr)
          pv[(size_t)(tile0 + h * 8 + rr) * 192 + (t * 16 + nlo) * 3 + i] = (bf16)acc[rr];
      }
    }
  }

  // ---- gate: f0 = [xs, |xv|] staged to LDS ----
  {
    const int r = lane >> 1, half = lane & 1;
    const float* xr = x + (size_t)(tile0 + r) * 256;
#pragma unroll
    for (int j = 0; j < 32; ++j) {
      int u = half * 32 + j;
      sb[wv][r][u] = (bf16)xr[u];
      float vx = xr[64 + 3 * u], vy = xr[64 + 3 * u + 1], vz = xr[64 + 3 * u + 2];
      sb[wv][r][64 + u] = (bf16)sqrtf(vx * vx + vy * vy + vz * vz + 1e-12f);
    }
  }
  lds_fence();
  v16bf f0[4];
#pragma unroll
  for (int kc = 0; kc < 4; ++kc) f0[kc] = load_a_bf16row(&sb[wv][nlo][0], kc * 32);
  // g1 = silu(f0 @ Wg1 + bg1) -> back into sb
#pragma unroll
  for (int t = 0; t < 8; ++t) {
    v8f acc = {};
#pragma unroll
    for (int kc = 0; kc < 4; ++kc)
      acc = wmma_bf16(f0[kc], load_bp(Wg1, 4, kc, t), acc);
    float bv = bg1[t * 16 + nlo];
#pragma unroll
    for (int rr = 0; rr < 8; ++rr)
      sb[wv][h * 8 + rr][t * 16 + nlo] = (bf16)siluf(acc[rr] + bv);
  }
  lds_fence();
  v16bf g1[4];
#pragma unroll
  for (int kc = 0; kc < 4; ++kc) g1[kc] = load_a_bf16row(&sb[wv][nlo][0], kc * 32);
  // g = g1 @ Wg2 + bg2 -> sg (f32)
#pragma unroll
  for (int t = 0; t < 8; ++t) {
    v8f acc = {};
#pragma unroll
    for (int kc = 0; kc < 4; ++kc)
      acc = wmma_bf16(g1[kc], load_bp(Wg2, 4, kc, t), acc);
    float bv = bg2[t * 16 + nlo];
#pragma unroll
    for (int rr = 0; rr < 8; ++rr)
      sg[wv][h * 8 + rr][t * 16 + nlo] = acc[rr] + bv;
  }
  lds_fence();

  // ---- xg scalar: gs = g[:, :64] @ (Wn0/8) + bn0 ----
  v16bf gaf0 = load_a_f32row(&sg[wv][nlo][0], 0);
  v16bf gaf1 = load_a_f32row(&sg[wv][nlo][0], 32);
#pragma unroll
  for (int t = 0; t < 4; ++t) {
    v8f acc = {};
    acc = wmma_bf16(gaf0, load_bp(Wn0, 2, 0, t), acc);
    acc = wmma_bf16(gaf1, load_bp(Wn0, 2, 1, t), acc);
    float bv = bn0[t * 16 + nlo];
#pragma unroll
    for (int rr = 0; rr < 8; ++rr)
      gs[(size_t)(tile0 + h * 8 + rr) * 64 + t * 16 + nlo] = acc[rr] + bv;
  }

  // ---- xg vector: gv[:,:,i] = (xv[:,:,i] * g[:,64+u]) @ (Wn1/8) ----
  {
    v16bf af[3][2];
    const float* xv = xrow + 64;
#pragma unroll
    for (int kc2 = 0; kc2 < 2; ++kc2) {
      const int kc = kc2 * 32;
      Span24 lo, hi;
      load_span24(xv + 3 * (kc + h * 8), lo);
      load_span24(xv + 3 * (kc + 16 + h * 8), hi);
      // gate values for the two K-subranges (vector loads from LDS)
      union { v4f v[2]; float f[8]; } glo, ghi;
      glo.v[0] = *(const v4f*)&sg[wv][nlo][64 + kc + h * 8];
      glo.v[1] = *(const v4f*)&sg[wv][nlo][64 + kc + h * 8 + 4];
      ghi.v[0] = *(const v4f*)&sg[wv][nlo][64 + kc + 16 + h * 8];
      ghi.v[1] = *(const v4f*)&sg[wv][nlo][64 + kc + 16 + h * 8 + 4];
#pragma unroll
      for (int i = 0; i < 3; ++i)
#pragma unroll
        for (int e = 0; e < 8; ++e) {
          af[i][kc2][e] = (bf16)(lo.f[3 * e + i] * glo.f[e]);
          af[i][kc2][8 + e] = (bf16)(hi.f[3 * e + i] * ghi.f[e]);
        }
    }
#pragma unroll 1
    for (int i = 0; i < 3; ++i) {
#pragma unroll
      for (int t = 0; t < 4; ++t) {
        v8f acc = {};
        acc = wmma_bf16(af[i][0], load_bp(Wn1, 2, 0, t), acc);
        acc = wmma_bf16(af[i][1], load_bp(Wn1, 2, 1, t), acc);
#pragma unroll
        for (int rr = 0; rr < 8; ++rr)
          gv[(size_t)(tile0 + h * 8 + rr) * 192 + (t * 16 + nlo) * 3 + i] = acc[rr];
      }
    }
  }
}

// ---------- edge kernel: fcnets + tensor product + scatter-add ----------

__global__ void __launch_bounds__(128) edge_kernel(
    const int* __restrict__ eidx, const float* __restrict__ edge_sh,
    const float* __restrict__ edge_attr,
    const bf16* __restrict__ ps, const bf16* __restrict__ pv,
    const float* __restrict__ gs, const float* __restrict__ gv,
    const bf16* __restrict__ Wf1, const bf16* __restrict__ Wf2,
    const bf16* __restrict__ Wl1, const bf16* __restrict__ Wl2,
    float* __restrict__ acc_out) {
  __shared__ alignas(16) bf16  s0b[4][16][192];  // [ps_dst | ps_dst | ip1]
  __shared__ alignas(16) float hbuf[4][16][64];  // h1 cols 0..31, h2 cols 32..63
  __shared__ int   meta[4][2][16];               // dst, src
  __shared__ float shs[4][16][4];                // edge_sh
  const int wv = threadIdx.x >> 5;
  const int lane = threadIdx.x & 31;
  const int e0 = (blockIdx.x * 4 + wv) * 16;
  const int nlo = lane & 15, h = lane >> 4;

  // ---- gather phase: 2 lanes per edge ----
  {
    const int r = lane >> 1, half = lane & 1;
    const int e = e0 + r;
    const int dst = eidx[e], src = eidx[N_EDGES + e];
    if (half == 0) {
      meta[wv][0][r] = dst;
      meta[wv][1][r] = src;
    } else {
#pragma unroll
      for (int c2 = 0; c2 < 4; ++c2) shs[wv][r][c2] = edge_sh[(size_t)e * 4 + c2];
    }
    const int u0 = half * 32;
    // ps row: 4x v8bf vector copies, duplicated into [0..63] and [64..127]
    const v8bf* pr8 = (const v8bf*)(ps + (size_t)dst * 64 + u0);
#pragma unroll
    for (int q = 0; q < 4; ++q) {
      v8bf vq = pr8[q];
      *(v8bf*)&s0b[wv][r][u0 + q * 8] = vq;
      *(v8bf*)&s0b[wv][r][64 + u0 + q * 8] = vq;
    }
    // ip1: 96 contiguous bf16 per row-half, vector loads into unions
    const v8bf* pd8 = (const v8bf*)(pv + (size_t)dst * 192 + u0 * 3);
    const v8bf* pp8 = (const v8bf*)(pv + (size_t)src * 192 + u0 * 3);
    union { v8bf v[12]; bf16 s[96]; } db, pb;
#pragma unroll
    for (int q = 0; q < 12; ++q) {
      db.v[q] = pd8[q];
      pb.v[q] = pp8[q];
    }
#pragma unroll
    for (int j = 0; j < 32; ++j) {
      float d3 = 0.f;
#pragma unroll
      for (int i = 0; i < 3; ++i)
        d3 += (float)db.s[j * 3 + i] * (float)pb.s[j * 3 + i];
      s0b[wv][r][128 + u0 + j] = (bf16)(d3 * (1.f / 3.f));
    }
  }
  lds_fence();

  // ---- h1 = ssp(edge_attr @ (Wf1/8)) ----
  const float* earow = edge_attr + (size_t)(e0 + nlo) * 64;
  v16bf ea0 = load_a_f32row(earow, 0);
  v16bf ea1 = load_a_f32row(earow, 32);
#pragma unroll
  for (int t = 0; t < 2; ++t) {
    v8f a = {};
    a = wmma_bf16(ea0, load_bp(Wf1, 2, 0, t), a);
    a = wmma_bf16(ea1, load_bp(Wf1, 2, 1, t), a);
#pragma unroll
    for (int rr = 0; rr < 8; ++rr)
      hbuf[wv][h * 8 + rr][t * 16 + nlo] = sspf(a[rr]);
  }
  // ---- h2 = ssp(s0 @ (Wl1/sqrt(192))) ----
  v16bf sfr[6];
#pragma unroll
  for (int kc = 0; kc < 6; ++kc)
    sfr[kc] = load_a_bf16row(&s0b[wv][nlo][0], kc * 32);
#pragma unroll
  for (int t = 0; t < 2; ++t) {
    v8f a = {};
#pragma unroll
    for (int kc = 0; kc < 6; ++kc)
      a = wmma_bf16(sfr[kc], load_bp(Wl1, 6, kc, t), a);
#pragma unroll
    for (int rr = 0; rr < 8; ++rr)
      hbuf[wv][h * 8 + rr][32 + t * 16 + nlo] = sspf(a[rr]);
  }
  lds_fence();
  v16bf h1 = load_a_f32row(&hbuf[wv][nlo][0], 0);
  v16bf h2 = load_a_f32row(&hbuf[wv][nlo][32], 0);

  // ---- per 16-feature tile: w = wf*wl, tensor product, scatter-add ----
#pragma unroll 1
  for (int ct = 0; ct < 4; ++ct) {
    v8f w[5];
#pragma unroll
    for (int j = 0; j < 5; ++j) {
      v8f zf = {};
      v8f wf = wmma_bf16(h1, load_bp(Wf2, 1, 0, j * 4 + ct), zf);
      v8f wl = wmma_bf16(h2, load_bp(Wl2, 1, 0, j * 4 + ct), zf);
#pragma unroll
      for (int rr = 0; rr < 8; ++rr) w[j][rr] = wf[rr] * wl[rr];
    }
    const int feat = ct * 16 + nlo;
#pragma unroll
    for (int rr = 0; rr < 8; ++rr) {
      int er = h * 8 + rr;
      int dst = meta[wv][0][er], src = meta[wv][1][er];
      float sh0v = shs[wv][er][0];
      float s1x = shs[wv][er][1], s1y = shs[wv][er][2], s1z = shs[wv][er][3];
      float es = gs[(size_t)src * 64 + feat];
      const float* evp = gv + (size_t)src * 192 + feat * 3;
      float evx = evp[0], evy = evp[1], evz = evp[2];
      float dv = evx * s1x + evy * s1y + evz * s1z;
      float outs = 0.44721359549995794f * (w[0][rr] * es * sh0v +
                                           w[3][rr] * dv * 0.57735026918962576f);
      float cx = evy * s1z - evz * s1y;
      float cy = evz * s1x - evx * s1z;
      float cz = evx * s1y - evy * s1x;
      float k1 = w[1][rr] * es, k2 = w[2][rr] * sh0v,
            k4 = w[4][rr] * 0.70710678118654752f;
      float ovx = 0.77459666924148338f * (k1 * s1x + k2 * evx + k4 * cx);
      float ovy = 0.77459666924148338f * (k1 * s1y + k2 * evy + k4 * cy);
      float ovz = 0.77459666924148338f * (k1 * s1z + k2 * evz + k4 * cz);
      float* ap = acc_out + (size_t)dst * 256;
      unsafeAtomicAdd(ap + feat, outs);
      unsafeAtomicAdd(ap + 64 + feat * 3 + 0, ovx);
      unsafeAtomicAdd(ap + 64 + feat * 3 + 1, ovy);
      unsafeAtomicAdd(ap + 64 + feat * 3 + 2, ovz);
    }
  }
}

// ---------- output kernel: out = e3_linear(acc + self_x) in-place ----------

__global__ void __launch_bounds__(64) out_kernel(
    float* out, const float* __restrict__ gs, const float* __restrict__ gv,
    const bf16* __restrict__ Wo0, const float* __restrict__ bo0,
    const bf16* __restrict__ Wo1) {
  const int wv = threadIdx.x >> 5;
  const int lane = threadIdx.x & 31;
  const int tile0 = (blockIdx.x * 2 + wv) * 16;
  const int nlo = lane & 15, h = lane >> 4;

  // load ALL fragments before any store (in-place transform, safe within wave)
  const float* orow = out + (size_t)(tile0 + nlo) * 256;
  const float* gsrow = gs + (size_t)(tile0 + nlo) * 64;
  const float* gvrow = gv + (size_t)(tile0 + nlo) * 192;
  v16bf as0 = load_a_f32sum(orow, gsrow, 0);
  v16bf as1 = load_a_f32sum(orow, gsrow, 32);
  v16bf av[3][2];
#pragma unroll
  for (int kc2 = 0; kc2 < 2; ++kc2) {
    const int kc = kc2 * 32;
    Span24 alo, ahi, blo, bhi;
    load_span24(orow + 64 + 3 * (kc + h * 8), alo);
    load_span24(orow + 64 + 3 * (kc + 16 + h * 8), ahi);
    load_span24(gvrow + 3 * (kc + h * 8), blo);
    load_span24(gvrow + 3 * (kc + 16 + h * 8), bhi);
#pragma unroll
    for (int i = 0; i < 3; ++i)
#pragma unroll
      for (int e = 0; e < 8; ++e) {
        av[i][kc2][e] = (bf16)(alo.f[3 * e + i] + blo.f[3 * e + i]);
        av[i][kc2][8 + e] = (bf16)(ahi.f[3 * e + i] + bhi.f[3 * e + i]);
      }
  }

#pragma unroll
  for (int t = 0; t < 4; ++t) {
    v8f acc = {};
    acc = wmma_bf16(as0, load_bp(Wo0, 2, 0, t), acc);
    acc = wmma_bf16(as1, load_bp(Wo0, 2, 1, t), acc);
    float bv = bo0[t * 16 + nlo];
#pragma unroll
    for (int rr = 0; rr < 8; ++rr)
      out[(size_t)(tile0 + h * 8 + rr) * 256 + t * 16 + nlo] = acc[rr] + bv;
  }
#pragma unroll 1
  for (int i = 0; i < 3; ++i) {
#pragma unroll
    for (int t = 0; t < 4; ++t) {
      v8f acc = {};
      acc = wmma_bf16(av[i][0], load_bp(Wo1, 2, 0, t), acc);
      acc = wmma_bf16(av[i][1], load_bp(Wo1, 2, 1, t), acc);
#pragma unroll
      for (int rr = 0; rr < 8; ++rr)
        out[(size_t)(tile0 + h * 8 + rr) * 256 + 64 + (t * 16 + nlo) * 3 + i] = acc[rr];
    }
  }
}

// ---------- launch ----------

extern "C" void kernel_launch(void* const* d_in, const int* in_sizes, int n_in,
                              void* d_out, int out_size, void* d_ws, size_t ws_size,
                              hipStream_t stream) {
  const float* x      = (const float*)d_in[0];
  const int*   eidx   = (const int*)d_in[1];
  const float* esh    = (const float*)d_in[2];
  const float* eattr  = (const float*)d_in[3];
  const float* W_pre0 = (const float*)d_in[4];
  const float* b_pre0 = (const float*)d_in[5];
  const float* W_pre1 = (const float*)d_in[6];
  const float* Wg1    = (const float*)d_in[7];
  const float* bg1    = (const float*)d_in[8];
  const float* Wg2    = (const float*)d_in[9];
  const float* bg2    = (const float*)d_in[10];
  const float* Wn0    = (const float*)d_in[11];
  const float* bn0    = (const float*)d_in[12];
  const float* Wn1    = (const float*)d_in[13];
  const float* Wf1    = (const float*)d_in[14];
  const float* Wf2    = (const float*)d_in[15];
  const float* Wl1    = (const float*)d_in[16];
  const float* Wl2    = (const float*)d_in[17];
  const float* Wo0    = (const float*)d_in[18];
  const float* bo0    = (const float*)d_in[19];
  const float* Wo1    = (const float*)d_in[20];

  char* ws = (char*)d_ws;
  bf16* arena = (bf16*)ws;
  bf16* bWpre0 = arena + 0;
  bf16* bWpre1 = arena + 4096;
  bf16* bWg1   = arena + 8192;
  bf16* bWg2   = arena + 24576;
  bf16* bWn0   = arena + 40960;
  bf16* bWn1   = arena + 45056;
  bf16* bWf1   = arena + 49152;
  bf16* bWf2   = arena + 51200;
  bf16* bWl1   = arena + 61440;
  bf16* bWl2   = arena + 67584;
  bf16* bWo0   = arena + 77824;
  bf16* bWo1   = arena + 81920;
  bf16*  ps = (bf16*)(ws + 172032);     // N*64  bf16
  bf16*  pv = (bf16*)(ws + 2732032);    // N*192 bf16
  float* gs = (float*)(ws + 10412032);  // N*64  f32
  float* gv = (float*)(ws + 15532032);  // N*192 f32  (ends ~30.9 MB)

  const float SM   = 0.125f;  // 1/sqrt(64)
  const float S32  = 0.17677669529663688f;
  const float S192 = 0.07216878364870322f;

  auto pk = [&](bf16* d, const float* s, int K, int N, float sc) {
    pack_b<<<(K * N + 255) / 256, 256, 0, stream>>>(d, s, K, N, sc);
  };
  pk(bWpre0, W_pre0, 64, 64, SM);
  pk(bWpre1, W_pre1, 64, 64, SM);
  pk(bWg1, Wg1, 128, 128, 1.f);
  pk(bWg2, Wg2, 128, 128, 1.f);
  pk(bWn0, Wn0, 64, 64, SM);
  pk(bWn1, Wn1, 64, 64, SM);
  pk(bWf1, Wf1, 64, 32, SM);
  pk(bWf2, Wf2, 32, 320, S32);
  pk(bWl1, Wl1, 192, 32, S192);
  pk(bWl2, Wl2, 32, 320, S32);
  pk(bWo0, Wo0, 64, 64, SM);
  pk(bWo1, Wo1, 64, 64, SM);

  hipMemsetAsync(d_out, 0, (size_t)out_size * sizeof(float), stream);

  node_kernel<<<625, 64, 0, stream>>>(x, bWpre0, b_pre0, bWpre1, bWg1, bg1, bWg2,
                                      bg2, bWn0, bn0, bWn1, ps, pv, gs, gv);
  edge_kernel<<<5000, 128, 0, stream>>>(eidx, esh, eattr, ps, pv, gs, gv, bWf1,
                                        bWf2, bWl1, bWl2, (float*)d_out);
  out_kernel<<<625, 64, 0, stream>>>((float*)d_out, gs, gv, bWo0, bo0, bWo1);
}
